// WindowAttention_5677946766063
// MI455X (gfx1250) — matrix operand order repaired and checked by reading, not compiled, not run on hardware
//
#include <hip/hip_runtime.h>
#include <hip/hip_bf16.h>

// Fully-fused Swin window attention for gfx1250 (MI455X).
// One workgroup (512 threads = 16 wave32) per window. All intermediates in LDS.
// f16 WMMA (v_wmma_f32_16x16x32_f16) for all four matmul stages.
// A prep kernel pre-converts weights to f16 and materializes the padded
// rel-pos bias so the hot kernel does no f32->f16 weight conversion.

typedef _Float16 half_t;
typedef __attribute__((ext_vector_type(16))) _Float16 v16h;
typedef __attribute__((ext_vector_type(8)))  _Float16 v8h;
typedef __attribute__((ext_vector_type(4)))  _Float16 v4h;
typedef __attribute__((ext_vector_type(8)))  float    v8f;

#define WMMA_F16(A, B, C) \
    __builtin_amdgcn_wmma_f32_16x16x32_f16(false, (A), false, (B), (short)0, (C), false, false)

static __device__ inline v16h cat8(v8h a, v8h b) {
    return __builtin_shufflevector(a, b, 0,1,2,3,4,5,6,7,8,9,10,11,12,13,14,15);
}

// B fragment from 16 consecutive f16 (pre-converted weights / LDS rows).
// B layout: lanes 0-15 hold K=0..15, lanes 16-31 hold K=16..31; elem j = B[kb+j][n].
static __device__ inline v16h load_h16(const half_t* __restrict__ p) {
    const v8h* q = (const v8h*)p;
    return cat8(q[0], q[1]);
}

// Fallback: B fragment from 16 consecutive f32 weights (convert on the fly).
static __device__ inline v16h load_w16(const float* __restrict__ p) {
    v16h r;
    const float4* q = (const float4*)p;
#pragma unroll
    for (int i = 0; i < 4; ++i) {
        float4 f = q[i];
        r[4*i + 0] = (_Float16)f.x;
        r[4*i + 1] = (_Float16)f.y;
        r[4*i + 2] = (_Float16)f.z;
        r[4*i + 3] = (_Float16)f.w;
    }
    return r;
}

// A fragment (16x32 f16): lane l<16 row m holds K base..base+15 (elems 0-15);
// lane l>=16 holds K base+16..base+31 (matches B contiguous order).
static __device__ inline v16h load_a_frag(const half_t* __restrict__ row, int hi) {
    v8h lo = *(const v8h*)(row + hi * 16);
    v8h hp = *(const v8h*)(row + hi * 16 + 8);
    return cat8(lo, hp);
}

#define NWIN   4096
#define NTOK   49
#define NPAD   64
#define CDIM   128
#define NHEAD  4
#define HDIM   32
#define QSCALE 0.17677669529663687f  // 32^-0.5

#define WQ_ELEMS   (384 * 128)
#define WP_ELEMS   (128 * 128)
#define BIAS_ELEMS (NHEAD * NPAD * NPAD)
#define PREP_TOTAL (WQ_ELEMS + WP_ELEMS + BIAS_ELEMS)

// ---------------------------------------------------------------------------
// Prep: qkv_w/proj_w f32 -> f16 (row-major, unchanged layout), and padded
// bias[h][64][64] f32 with -1e30 outside the valid 49x49 window.
// ---------------------------------------------------------------------------
__global__ __launch_bounds__(512)
void prep_kernel(const float* __restrict__ qkv_w, const float* __restrict__ proj_w,
                 const float* __restrict__ rpb_table, const int* __restrict__ rpi,
                 half_t* __restrict__ wq_h, half_t* __restrict__ wp_h,
                 float* __restrict__ bias_pad) {
    for (int i = blockIdx.x * blockDim.x + threadIdx.x; i < PREP_TOTAL;
         i += gridDim.x * blockDim.x) {
        if (i < WQ_ELEMS) {
            wq_h[i] = (half_t)qkv_w[i];
        } else if (i < WQ_ELEMS + WP_ELEMS) {
            int j = i - WQ_ELEMS;
            wp_h[j] = (half_t)proj_w[j];
        } else {
            int j = i - (WQ_ELEMS + WP_ELEMS);       // [h][m][n]
            int h = j >> 12, m = (j >> 6) & 63, n = j & 63;
            float bv = -1e30f;
            if (m < NTOK && n < NTOK) bv = rpb_table[rpi[m * NTOK + n] * NHEAD + h];
            bias_pad[j] = bv;
        }
    }
}

// ---------------------------------------------------------------------------
// Main fused kernel. PREP=true uses f16 weights + padded bias from workspace.
// ---------------------------------------------------------------------------
template <bool PREP>
__global__ __launch_bounds__(512)
void window_attn_fused(const float* __restrict__ x,
                       const float* __restrict__ qkv_w,
                       const float* __restrict__ qkv_b,
                       const float* __restrict__ proj_w,
                       const float* __restrict__ proj_b,
                       const float* __restrict__ rpb_table,
                       const int*   __restrict__ rpi,
                       const half_t* __restrict__ wq_h,
                       const half_t* __restrict__ wp_h,
                       const float*  __restrict__ bias_pad,
                       float* __restrict__ out) {
    // 64KB LDS, phase-overlapped regions:
    //  [ 0,16K): x tile (phase 0)          -> attn probs low half  (phase 2+)
    //  [16,32K): q f16 [4][64][32]         -> attn probs high half (phase 2+)
    //  [32,48K): k f16 [4][64][32]         -> attn-out tile [64][128] (phase 3+)
    //  [48,64K): v^T f16 [4][32][64]
    __shared__ char smem[65536];
    half_t* x_s    = (half_t*)smem;               // [64][128]
    half_t* q_s    = (half_t*)(smem + 16384);     // [4][64][32]
    half_t* k_s    = (half_t*)(smem + 32768);     // [4][64][32]
    half_t* vt_s   = (half_t*)(smem + 49152);     // [4][32][64]
    half_t* attn_s = (half_t*)smem;               // [4][64][64]
    half_t* out_s  = (half_t*)(smem + 32768);     // [64][128]

    const int b    = blockIdx.x;
    const int tid  = threadIdx.x;
    const int lane = tid & 31;
    // Force the wave id (and everything derived from it) into SGPRs.
    const int wave = __builtin_amdgcn_readfirstlane(tid >> 5);   // 0..15
    const int l15  = lane & 15;
    const int hi   = lane >> 4;    // which 16-lane half

    // ---- Phase 0: stage x[b] (49x128 f32) -> x_s (64x128 f16, zero-padded)
    for (int i = tid; i < NPAD * (CDIM / 4); i += 512) {
        int row = i >> 5, c4 = i & 31;
        v4h hv = {};
        if (row < NTOK) {
            float4 f = ((const float4*)(x + ((size_t)b * NTOK + row) * CDIM))[c4];
            hv[0] = (_Float16)f.x; hv[1] = (_Float16)f.y;
            hv[2] = (_Float16)f.z; hv[3] = (_Float16)f.w;
        }
        *(v4h*)&x_s[row * CDIM + c4 * 4] = hv;
    }
    __syncthreads();

    const int mtile = wave & 3;    // 4 row tiles of 16 (scalar)
    const int ngrp  = wave >> 2;   // 0..3 (scalar)

    // ---- Phase 1: QKV GEMM. Each wave: mtile x 6 column tiles (24 total = 384 cols)
    {
        const int m = mtile * 16 + l15;
        v16h afrag[4];
#pragma unroll
        for (int ks = 0; ks < 4; ++ks)
            afrag[ks] = load_a_frag(&x_s[m * CDIM + ks * 32], hi);

        for (int nt = 0; nt < 6; ++nt) {
            const int ntile = ngrp * 6 + nt;          // scalar
            const int o = ntile * 16 + l15;           // 0..383
            v8f acc = {};
#pragma unroll
            for (int ks = 0; ks < 4; ++ks) {
                v16h bfrag;
                if (PREP) bfrag = load_h16(&wq_h[(size_t)o * CDIM + ks * 32 + hi * 16]);
                else      bfrag = load_w16(&qkv_w[(size_t)o * CDIM + ks * 32 + hi * 16]);
                acc = WMMA_F16(afrag[ks], bfrag, acc);
            }
            const float bias = qkv_b[o];
            // Tile-uniform (scalar) decode: 16-wide tiles never straddle the
            // q/k/v (128) or head (32) boundaries.
            const int g     = ntile >> 3;        // 0=q 1=k 2=v   (scalar)
            const int h2    = (ntile >> 1) & 3;  // head          (scalar)
            const int dbase = (ntile & 1) * 16;  //               (scalar)
            const int d     = dbase + l15;       // dim within head
            const int mrow  = mtile * 16 + hi * 8;
            if (g == 0) {
                half_t* dst = &q_s[(h2 * NPAD + mrow) * HDIM + d];
#pragma unroll
                for (int r = 0; r < 8; ++r)
                    dst[r * HDIM] = (half_t)((acc[r] + bias) * QSCALE);
            } else if (g == 1) {
                half_t* dst = &k_s[(h2 * NPAD + mrow) * HDIM + d];
#pragma unroll
                for (int r = 0; r < 8; ++r)
                    dst[r * HDIM] = (half_t)(acc[r] + bias);
            } else {
                half_t* dst = &vt_s[(h2 * HDIM + d) * NPAD + mrow];  // transposed
#pragma unroll
                for (int r = 0; r < 8; ++r)
                    dst[r] = (half_t)(acc[r] + bias);
            }
        }
    }
    __syncthreads();

    // ---- Phase 2: Q@K^T + bias + softmax. Wave -> (head, mtile) strip 16x64.
    const int h = wave >> 2;  // scalar head id
    v8f acc[4];
    {
        const int m = mtile * 16 + l15;
        v16h aq = load_a_frag(&q_s[(h * NPAD + m) * HDIM], hi);  // k-dim = 32 = one step
#pragma unroll
        for (int nt = 0; nt < 4; ++nt) {
            const int n = nt * 16 + l15;
            // B = K^T (d x n): column n of B = row n of K -> contiguous d
            v16h bf = load_h16(&k_s[(h * NPAD + n) * HDIM + hi * 16]);
            v8f c = {};
            acc[nt] = WMMA_F16(aq, bf, c);
        }
    }
    // relative-position bias + padding mask
#pragma unroll
    for (int nt = 0; nt < 4; ++nt) {
        const int n = nt * 16 + l15;
#pragma unroll
        for (int r = 0; r < 8; ++r) {
            const int m = mtile * 16 + r + hi * 8;
            float bv;
            if (PREP) {
                bv = bias_pad[(h * NPAD + m) * NPAD + n];
            } else {
                bv = -1e30f;
                if (m < NTOK && n < NTOK) bv = rpb_table[rpi[m * NTOK + n] * NHEAD + h];
            }
            acc[nt][r] += bv;
        }
    }
    // row softmax: each lane holds 4 cols of a row; 16-lane half holds all 64.
#pragma unroll
    for (int r = 0; r < 8; ++r) {
        float mx = fmaxf(fmaxf(acc[0][r], acc[1][r]), fmaxf(acc[2][r], acc[3][r]));
#pragma unroll
        for (int s = 1; s <= 8; s <<= 1) mx = fmaxf(mx, __shfl_xor(mx, s, 32));
        float sum = 0.0f;
#pragma unroll
        for (int nt = 0; nt < 4; ++nt) {
            float e = __expf(acc[nt][r] - mx);
            acc[nt][r] = e;
            sum += e;
        }
#pragma unroll
        for (int s = 1; s <= 8; s <<= 1) sum += __shfl_xor(sum, s, 32);
        const float inv = (sum > 0.0f) ? 1.0f / sum : 0.0f;
#pragma unroll
        for (int nt = 0; nt < 4; ++nt) acc[nt][r] *= inv;
    }
    __syncthreads();  // all q/k LDS reads complete before attn_s overwrites them
    {
        const int mrow = mtile * 16 + hi * 8;
#pragma unroll
        for (int nt = 0; nt < 4; ++nt) {
            half_t* dst = &attn_s[(h * NPAD + mrow) * NPAD + nt * 16 + l15];
#pragma unroll
            for (int r = 0; r < 8; ++r)
                dst[r * NPAD] = (half_t)acc[nt][r];
        }
    }
    __syncthreads();

    // ---- Phase 3: attn @ V -> out_s strip 16x32 at columns [h*32, h*32+32)
    {
        const int m = mtile * 16 + l15;
        v16h afr[2];
#pragma unroll
        for (int ks = 0; ks < 2; ++ks)
            afr[ks] = load_a_frag(&attn_s[(h * NPAD + m) * NPAD + ks * 32], hi);
#pragma unroll
        for (int dt = 0; dt < 2; ++dt) {
            const int d = dt * 16 + l15;
            v8f c = {};
#pragma unroll
            for (int ks = 0; ks < 2; ++ks) {
                // B = V (token x d): column d of B contiguous in vt_s
                v16h bf = load_h16(&vt_s[(h * HDIM + d) * NPAD + ks * 32 + hi * 16]);
                c = WMMA_F16(afr[ks], bf, c);
            }
            half_t* dst = &out_s[(mtile * 16 + hi * 8) * CDIM + h * HDIM + dt * 16 + l15];
#pragma unroll
            for (int r = 0; r < 8; ++r)
                dst[r * CDIM] = (half_t)c[r];
        }
    }
    __syncthreads();

    // ---- Phase 4: projection. 4 mtiles x 8 otiles = 32 tiles, 2 per wave.
    {
        const int m = mtile * 16 + l15;
        v16h afr[4];
#pragma unroll
        for (int ks = 0; ks < 4; ++ks)
            afr[ks] = load_a_frag(&out_s[m * CDIM + ks * 32], hi);
        const int og = wave >> 2;  // scalar: otiles {2og, 2og+1}
#pragma unroll
        for (int ot = 0; ot < 2; ++ot) {
            const int o = (og * 2 + ot) * 16 + l15;  // 0..127
            v8f c = {};
#pragma unroll
            for (int ks = 0; ks < 4; ++ks) {
                v16h bf;
                if (PREP) bf = load_h16(&wp_h[(size_t)o * CDIM + ks * 32 + hi * 16]);
                else      bf = load_w16(&proj_w[(size_t)o * CDIM + ks * 32 + hi * 16]);
                c = WMMA_F16(afr[ks], bf, c);
            }
            const float pb = proj_b[o];
#pragma unroll
            for (int r = 0; r < 8; ++r) {
                const int mm = mtile * 16 + r + hi * 8;
                if (mm < NTOK)
                    out[((size_t)b * NTOK + mm) * CDIM + o] = c[r] + pb;
            }
        }
    }
}

extern "C" void kernel_launch(void* const* d_in, const int* in_sizes, int n_in,
                              void* d_out, int out_size, void* d_ws, size_t ws_size,
                              hipStream_t stream) {
    (void)in_sizes; (void)n_in; (void)out_size;
    const float* x         = (const float*)d_in[0];
    const float* qkv_w     = (const float*)d_in[1];
    const float* qkv_b     = (const float*)d_in[2];
    const float* proj_w    = (const float*)d_in[3];
    const float* proj_b    = (const float*)d_in[4];
    const float* rpb_table = (const float*)d_in[5];
    const int*   rpi       = (const int*)d_in[6];
    float* out = (float*)d_out;

    const size_t need = (size_t)WQ_ELEMS * 2 + (size_t)WP_ELEMS * 2 +
                        (size_t)BIAS_ELEMS * 4;  // ~196.6 KB

    if (d_ws != nullptr && ws_size >= need) {
        half_t* wq_h = (half_t*)d_ws;
        half_t* wp_h = wq_h + WQ_ELEMS;
        float*  bias_pad = (float*)((char*)d_ws + (size_t)(WQ_ELEMS + WP_ELEMS) * 2);
        prep_kernel<<<160, 512, 0, stream>>>(qkv_w, proj_w, rpb_table, rpi,
                                             wq_h, wp_h, bias_pad);
        window_attn_fused<true><<<NWIN, 512, 0, stream>>>(
            x, qkv_w, qkv_b, proj_w, proj_b, rpb_table, rpi,
            wq_h, wp_h, bias_pad, out);
    } else {
        window_attn_fused<false><<<NWIN, 512, 0, stream>>>(
            x, qkv_w, qkv_b, proj_w, proj_b, rpb_table, rpi,
            nullptr, nullptr, nullptr, out);
    }
}
